// HybridAttentionTopKPool_61649960566956
// MI455X (gfx1250) — matrix operand order repaired and checked by reading, not compiled, not run on hardware
//
#include <hip/hip_runtime.h>
#include <hip/hip_bf16.h>
#include <stdint.h>

// ---- problem constants (from reference) ----
#define B_    64
#define S_    2048
#define D_    512
#define H_    128
#define Q_    4
#define QH_   32
#define TOPK_ 4
#define NEGV  (-1.0e9f)

#define W1T_BYTES (D_ * H_ * 2)   // 131072 B of bf16 weights

typedef __attribute__((ext_vector_type(16))) __bf16 v16bf;
typedef __attribute__((ext_vector_type(8)))  float  v8f;

// Native fptrunc -> v_cvt_pk_bf16_f32 on gfx1250 (verified in round 2 asm).
static __device__ __forceinline__ __bf16 f2bf(float f) {
  return static_cast<__bf16>(f);
}

// ---------------------------------------------------------------------------
// K0: convert W1 [D,H] fp32 -> W1t [H,D] bf16 (transposed so B-fragments are
// 16 contiguous bf16 = one 32-byte access).
// ---------------------------------------------------------------------------
__global__ void k0_convert_w1(const float* __restrict__ W1,
                              unsigned short* __restrict__ w1t) {
  int i = blockIdx.x * blockDim.x + threadIdx.x;     // 0 .. D*H-1
  if (i >= D_ * H_) return;
  int h = i % H_, d = i / H_;
  union { uint16_t u; __bf16 b; } c; c.b = f2bf(W1[(size_t)d * H_ + h]);
  w1t[(size_t)h * D_ + d] = c.u;
}

// ---------------------------------------------------------------------------
// K1: fused evidence GEMM (bf16 WMMA, fp32 accum) + tanh + w2 dot + quality
// MLP -> combined logits [B*S].
// W1t (128 KB bf16) is staged once per block into LDS; the 8 waves of the
// block then feed all B-fragments from LDS (ds_load) instead of re-reading
// 128 KB each from L2.  One wave handles 16 rows x H=128; K=512 -> 16 wmma
// k-steps x 8 n-tiles = 128 v_wmma_f32_16x16x32_bf16 per wave-tile.
// ---------------------------------------------------------------------------
__global__ __launch_bounds__(256) void k1_evidence(
    const float* __restrict__ feats, const unsigned short* __restrict__ w1t,
    const float* __restrict__ b1, const float* __restrict__ w2,
    const float* __restrict__ b2, const float* __restrict__ segq,
    const float* __restrict__ qW1, const float* __restrict__ qb1,
    const float* __restrict__ qw2, const float* __restrict__ qb2,
    float* __restrict__ combined) {
  extern __shared__ unsigned short lds_w1t[];        // 128 KB dynamic LDS

  // cooperative stage of W1t into LDS (256 threads x 32 uint4 = 128 KB)
  {
    const uint4* src = (const uint4*)w1t;
    uint4* dst = (uint4*)lds_w1t;
#pragma unroll
    for (int i = 0; i < (W1T_BYTES / 16) / 256; ++i)
      dst[threadIdx.x + i * 256] = src[threadIdx.x + i * 256];
  }
  __syncthreads();

  const int lane = threadIdx.x & 31;
  const int tile = blockIdx.x * (blockDim.x >> 5) + (threadIdx.x >> 5);
  const int row0 = tile * 16;                       // 16 rows of B*S
  const int half = lane >> 4;                       // 0 or 1
  const int lq   = lane & 15;

  v8f acc[8];
#pragma unroll
  for (int n = 0; n < 8; ++n) acc[n] = (v8f){0.f,0.f,0.f,0.f,0.f,0.f,0.f,0.f};

  const int arow = row0 + lq;                       // A row owned by this lane
  const float* arow_p = feats + (size_t)arow * D_;

  for (int ks = 0; ks < 16; ++ks) {
    const int kbase = ks * 32 + half * 16;          // A: K {kbase..+15}
    const float4* pa = (const float4*)(arow_p + kbase);
    float4 a0 = pa[0], a1 = pa[1];
    const float4* pb = (const float4*)(arow_p + kbase + 8);
    float4 a2 = pb[0], a3 = pb[1];
    if (ks + 1 < 16) __builtin_prefetch(arow_p + kbase + 32, 0, 1);

    v16bf A;
    A[0]=f2bf(a0.x); A[1]=f2bf(a0.y); A[2]=f2bf(a0.z); A[3]=f2bf(a0.w);
    A[4]=f2bf(a1.x); A[5]=f2bf(a1.y); A[6]=f2bf(a1.z); A[7]=f2bf(a1.w);
    A[8]=f2bf(a2.x); A[9]=f2bf(a2.y); A[10]=f2bf(a2.z); A[11]=f2bf(a2.w);
    A[12]=f2bf(a3.x); A[13]=f2bf(a3.y); A[14]=f2bf(a3.z); A[15]=f2bf(a3.w);

    const int kb2 = ks * 32 + half * 16;            // B: lanes 0-15 K 0..15, 16-31 K 16..31
#pragma unroll
    for (int n = 0; n < 8; ++n) {
      const v16bf Bf =
          *(const v16bf*)(lds_w1t + (size_t)(n * 16 + lq) * D_ + kb2);
      acc[n] = __builtin_amdgcn_wmma_f32_16x16x32_bf16(
          false, A, false, Bf, (short)0, acc[n], false, false);
    }
  }

  // epilogue: lane holds cols N = n*16+lq, rows M = v + half*8 (v = vgpr idx)
  float part[8];
#pragma unroll
  for (int v = 0; v < 8; ++v) {
    float s = 0.f;
#pragma unroll
    for (int n = 0; n < 8; ++n) {
      const int col = n * 16 + lq;
      s += tanhf(acc[n][v] + b1[col]) * w2[col];
    }
    part[v] = s;
  }
  // reduce across the 16 lanes of each half (xor masks < 16 stay in-half)
#pragma unroll
  for (int m = 1; m < 16; m <<= 1) {
#pragma unroll
    for (int v = 0; v < 8; ++v) part[v] += __shfl_xor(part[v], m, 32);
  }

  if (lq < 8) {
    const int r = row0 + half * 8 + lq;
    float ev = part[lq] + b2[0];
    // quality MLP: relu(q @ qW1 + qb1) @ qw2 + qb2
    float q0 = segq[(size_t)r * Q_ + 0], q1 = segq[(size_t)r * Q_ + 1];
    float q2 = segq[(size_t)r * Q_ + 2], q3 = segq[(size_t)r * Q_ + 3];
    float qs = qb2[0];
#pragma unroll
    for (int j = 0; j < QH_; ++j) {
      float hdn = qb1[j] + q0 * qW1[0 * QH_ + j] + q1 * qW1[1 * QH_ + j] +
                  q2 * qW1[2 * QH_ + j] + q3 * qW1[3 * QH_ + j];
      qs += fmaxf(hdn, 0.f) * qw2[j];
    }
    combined[r] = ev + 1.0f * qs;   // QUALITY_ALPHA = 1.0
  }
}

// ---------------------------------------------------------------------------
// K2: per batch row masked softmax over S + iterative top-4 argmax.
// ---------------------------------------------------------------------------
__global__ __launch_bounds__(256) void k2_softmax_topk(
    const float* __restrict__ combined, const unsigned char* __restrict__ mask,
    float* __restrict__ weights_out, int* __restrict__ idx_out,
    float* __restrict__ valid_out) {
  const int b = blockIdx.x, t = threadIdx.x;
  __shared__ float sc[S_];
  __shared__ float red[256];
  __shared__ int   redi[256];

  float lmax = -INFINITY;
  for (int i = t; i < S_; i += 256) {
    float v = mask[(size_t)b * S_ + i] ? combined[(size_t)b * S_ + i] : NEGV;
    sc[i] = v;
    lmax = fmaxf(lmax, v);
  }
  red[t] = lmax; __syncthreads();
  for (int off = 128; off > 0; off >>= 1) {
    if (t < off) red[t] = fmaxf(red[t], red[t + off]);
    __syncthreads();
  }
  const float m = red[0]; __syncthreads();

  float e[S_ / 256];
  float lsum = 0.f;
#pragma unroll
  for (int j = 0; j < S_ / 256; ++j) {
    int i = t + j * 256;
    float v = sc[i];
    float ex = (v > 0.5f * NEGV) ? __expf(v - m) : 0.f;
    e[j] = ex; lsum += ex;
  }
  red[t] = lsum; __syncthreads();
  for (int off = 128; off > 0; off >>= 1) {
    if (t < off) red[t] += red[t + off];
    __syncthreads();
  }
  const float denom = fmaxf(red[0], 1e-8f); __syncthreads();
#pragma unroll
  for (int j = 0; j < S_ / 256; ++j)
    weights_out[(size_t)b * S_ + t + j * 256] = e[j] / denom;

  // top-4: iterative block argmax over sc (ties -> lowest index, like top_k)
  for (int k = 0; k < TOPK_; ++k) {
    float bv = -INFINITY; int bi = 0x7fffffff;
    for (int i = t; i < S_; i += 256) {
      float v = sc[i];
      if (v > bv || (v == bv && i < bi)) { bv = v; bi = i; }
    }
    red[t] = bv; redi[t] = bi; __syncthreads();
    for (int off = 128; off > 0; off >>= 1) {
      if (t < off) {
        float v2 = red[t + off]; int i2 = redi[t + off];
        if (v2 > red[t] || (v2 == red[t] && i2 < redi[t])) { red[t] = v2; redi[t] = i2; }
      }
      __syncthreads();
    }
    if (t == 0) {
      int bi0 = redi[0];
      idx_out[b * TOPK_ + k] = bi0;
      valid_out[b * TOPK_ + k] = mask[(size_t)b * S_ + bi0] ? 1.f : 0.f;
      sc[bi0] = -INFINITY;
    }
    __syncthreads();
  }
}

// ---------------------------------------------------------------------------
// K3a: partial attention pooling, s split into 8 chunks for occupancy.
// ---------------------------------------------------------------------------
__global__ __launch_bounds__(256) void k3a_attn_partial(
    const float* __restrict__ feats, const float* __restrict__ weights,
    float* __restrict__ partial) {
  const int b = blockIdx.x, chunk = blockIdx.y, t = threadIdx.x;
  __shared__ float w[256];
  w[t] = weights[(size_t)b * S_ + chunk * 256 + t];
  __syncthreads();
  const float* fb = feats + ((size_t)b * S_ + (size_t)chunk * 256) * D_;
  for (int d = t; d < D_; d += 256) {
    float acc = 0.f;
#pragma unroll 4
    for (int s = 0; s < 256; ++s) acc += fb[(size_t)s * D_ + d] * w[s];
    partial[((size_t)b * 8 + chunk) * D_ + d] = acc;
  }
}

// ---------------------------------------------------------------------------
// K3b: finalize -> attn_pooled, topk_pooled, pooled.
// ---------------------------------------------------------------------------
__global__ __launch_bounds__(256) void k3b_finalize(
    const float* __restrict__ feats, const unsigned char* __restrict__ mask,
    const float* __restrict__ partial, const int* __restrict__ idx,
    float* __restrict__ pooled, float* __restrict__ attn,
    float* __restrict__ topkp) {
  const int b = blockIdx.x, t = threadIdx.x;
  __shared__ int   ti[TOPK_];
  __shared__ float tv[TOPK_];
  if (t < TOPK_) {
    int i = idx[b * TOPK_ + t];
    ti[t] = i;
    tv[t] = mask[(size_t)b * S_ + i] ? 1.f : 0.f;
  }
  __syncthreads();
  const float dn = fmaxf(tv[0] + tv[1] + tv[2] + tv[3], 1.f);
  for (int d = t; d < D_; d += 256) {
    float a = 0.f;
#pragma unroll
    for (int c = 0; c < 8; ++c) a += partial[((size_t)b * 8 + c) * D_ + d];
    float tk = 0.f;
#pragma unroll
    for (int k = 0; k < TOPK_; ++k)
      tk += tv[k] * feats[((size_t)b * S_ + ti[k]) * D_ + d];
    tk /= dn;
    attn[(size_t)b * D_ + d]   = a;
    topkp[(size_t)b * D_ + d]  = tk;
    pooled[(size_t)b * D_ + d] = 0.5f * a + 0.5f * tk;
  }
}

// ---------------------------------------------------------------------------
extern "C" void kernel_launch(void* const* d_in, const int* in_sizes, int n_in,
                              void* d_out, int out_size, void* d_ws, size_t ws_size,
                              hipStream_t stream) {
  const float*         feats = (const float*)d_in[0];
  const unsigned char* mask  = (const unsigned char*)d_in[1];
  const float*         segq  = (const float*)d_in[2];
  const float*         W1    = (const float*)d_in[3];
  const float*         b1    = (const float*)d_in[4];
  const float*         w2    = (const float*)d_in[5];
  const float*         b2    = (const float*)d_in[6];
  const float*         qW1   = (const float*)d_in[7];
  const float*         qb1   = (const float*)d_in[8];
  const float*         qw2   = (const float*)d_in[9];
  const float*         qb2   = (const float*)d_in[10];

  // outputs concatenated flat in return order
  float* outf      = (float*)d_out;
  float* pooled    = outf;                                  // [B,D]
  float* weights   = outf + (size_t)B_ * D_;                // [B,S]
  float* attn      = weights + (size_t)B_ * S_;             // [B,D]
  float* topkp     = attn + (size_t)B_ * D_;                // [B,D]
  int*   idx_out   = (int*)(topkp + (size_t)B_ * D_);       // [B,k] int32
  float* valid_out = (float*)(idx_out + (size_t)B_ * TOPK_);// [B,k]

  // workspace layout (~1.7 MB)
  char* ws = (char*)d_ws;
  unsigned short* w1t      = (unsigned short*)ws;                      // 128 KB
  float*          combined = (float*)(ws + W1T_BYTES);                 // 512 KB
  float*          partial  = (float*)(ws + W1T_BYTES + (size_t)B_*S_*4); // 1 MB

  // K0: W1 -> bf16 transposed
  k0_convert_w1<<<(D_ * H_ + 255) / 256, 256, 0, stream>>>(W1, w1t);

  // K1: WMMA evidence + quality -> combined logits (128 KB dynamic LDS)
  const int tiles = (B_ * S_) / 16;          // 8192 wave-tiles
  k1_evidence<<<tiles / 8, 256, W1T_BYTES, stream>>>(
      feats, w1t, b1, w2, b2, segq, qW1, qb1, qw2, qb2, combined);

  // K2: softmax + top-k
  k2_softmax_topk<<<B_, 256, 0, stream>>>(combined, mask, weights, idx_out,
                                          valid_out);

  // K3: pooling
  k3a_attn_partial<<<dim3(B_, 8), 256, 0, stream>>>(feats, weights, partial);
  k3b_finalize<<<B_, 256, 0, stream>>>(feats, mask, partial, idx_out,
                                       pooled, attn, topkp);
}
